// GatedAttentionUnit_50732153700661
// MI455X (gfx1250) — matrix operand order repaired and checked by reading, not compiled, not run on hardware
//
#include <hip/hip_runtime.h>

// Gated Attention Unit on gfx1250 (MI455X): all four einsums run on
// V_WMMA_F32_16X16X32_BF16 (bf16 in, f32 accumulate). Flash-style 2-pass
// softmax; P tile is re-laid-out through LDS between the QK^T WMMA (C layout)
// and the P@V WMMA (A layout).

#define HID   768
#define INTER 1536
#define DK    128
#define NB    4
#define SEQ   2048
#define BS    (NB * SEQ)          // 8192 rows total
#define NI    (2 * INTER + DK)    // 3200 output cols of GEMM1
#define LOG512F 6.2383246250395075f
#define RSQRT_DK 0.08838834764831845f  // 1/sqrt(128)

typedef __bf16 bf16;
typedef __attribute__((ext_vector_type(16))) __bf16 v16bf;
typedef __attribute__((ext_vector_type(8)))  __bf16 v8bf;
typedef __attribute__((ext_vector_type(8)))  float  v8f;

// ---------------------------------------------------------------------------
// WMMA fragment helpers (wave32). A: 16x32 bf16, rows on lanes, K in VGPRs.
//   lane half 0 (lanes 0-15):  row = lane,    K = 0..15
//   lane half 1 (lanes 16-31): row = lane-16, K = 16..31
// B: 32x16 bf16, column = lane%16; lanes 0-15 hold K=0..15, lanes 16-31 K=16..31
// (so a "Bt" matrix stored N-major with K contiguous loads as one v16bf).
// C/D: 8 VGPRs, col = lane%16, row r + 8*(lane/16), r = 0..7.
// ---------------------------------------------------------------------------

static __device__ __forceinline__ v16bf frag_a(const bf16* base, int ld, int m0, int k0) {
  const int lane = threadIdx.x & 31;
  const int half = lane >> 4, lr = lane & 15;
  const bf16* p = base + (size_t)(m0 + lr) * ld + k0 + half * 16;
  v8bf lo = *(const v8bf*)p;
  v8bf hi = *(const v8bf*)(p + 8);
  v16bf r;
#pragma unroll
  for (int i = 0; i < 8; ++i) { r[i] = lo[i]; r[8 + i] = hi[i]; }
  return r;
}

static __device__ __forceinline__ v16bf frag_bt(const bf16* base, int ld, int n0, int k0) {
  const int lane = threadIdx.x & 31;
  const int half = lane >> 4, lr = lane & 15;
  const bf16* p = base + (size_t)(n0 + lr) * ld + k0 + half * 16;
  return *(const v16bf*)p;
}

static __device__ __forceinline__ v8f wmma_bf16(v16bf a, v16bf b, v8f c) {
  return __builtin_amdgcn_wmma_f32_16x16x32_bf16(false, a, false, b, (short)0, c,
                                                 false, false);
}

// ---------------------------------------------------------------------------
// Prep kernels
// ---------------------------------------------------------------------------

__global__ void gau_cvt_bf16(const float* __restrict__ src, bf16* __restrict__ dst, int n) {
  int i = blockIdx.x * 256 + threadIdx.x;
  if (i < n) dst[i] = (bf16)src[i];
}

// src[rows][cols] (f32, row-major) -> dst[cols][rows] (bf16)
__global__ void gau_transpose_bf16(const float* __restrict__ src, bf16* __restrict__ dst,
                                   int rows, int cols) {
  int i = blockIdx.x * 256 + threadIdx.x;
  if (i < rows * cols) {
    int r = i / cols, c = i % cols;
    dst[(size_t)c * rows + r] = (bf16)src[i];
  }
}

// Per-batch softmax scale from attention mask: scl[2b] = log(l)/log512/sqrt(DK),
// scl[2b+1] = -1e12 * log(l)/log512 (pre-scaled masked score).
__global__ void gau_stats(const int* __restrict__ mask, float* __restrict__ scl) {
  __shared__ int red[256];
  const int b = blockIdx.x, t = threadIdx.x;
  int s = 0;
  for (int i = t; i < SEQ; i += 256) s += mask[b * SEQ + i];
  red[t] = s;
  __syncthreads();
  for (int o = 128; o > 0; o >>= 1) {
    if (t < o) red[t] += red[t + o];
    __syncthreads();
  }
  if (t == 0) {
    float l  = (float)(red[0] > 0 ? red[0] : 1);
    float ls = __logf(l) / LOG512F;
    scl[2 * b]     = ls * RSQRT_DK;
    scl[2 * b + 1] = -1.0e12f * ls;
  }
}

// ---------------------------------------------------------------------------
// GEMM1: X = silu(Hb @ WiT^T), split into U / Vt / Qb / Kb.  16x64 tile/wave.
// ---------------------------------------------------------------------------

__global__ __launch_bounds__(32) void gau_gemm1(
    const bf16* __restrict__ Hb, const bf16* __restrict__ WiT,
    const float* __restrict__ qg, const float* __restrict__ kg,
    bf16* __restrict__ U, bf16* __restrict__ Vt,
    bf16* __restrict__ Qb, bf16* __restrict__ Kb) {
  const int m0 = blockIdx.y * 16;
  const int n0 = blockIdx.x * 64;
  const int lane = threadIdx.x & 31, half = lane >> 4, lr = lane & 15;

  v8f acc[4] = {};
  for (int k0 = 0; k0 < HID; k0 += 32) {
    v16bf a = frag_a(Hb, HID, m0, k0);
#pragma unroll
    for (int j = 0; j < 4; ++j)
      acc[j] = wmma_bf16(a, frag_bt(WiT, HID, n0 + 16 * j, k0), acc[j]);
  }

#pragma unroll
  for (int j = 0; j < 4; ++j) {
    const int col = n0 + 16 * j + lr;
#pragma unroll
    for (int r = 0; r < 8; ++r) {
      const int m = m0 + r + 8 * half;
      float s = acc[j][r];
      s = s / (1.0f + __expf(-s));  // silu
      if (col < INTER) {
        U[(size_t)m * INTER + col] = (bf16)s;
      } else if (col < 2 * INTER) {
        Vt[(size_t)(col - INTER) * BS + m] = (bf16)s;  // V stored transposed
      } else {
        const int d = col - 2 * INTER;
        Qb[(size_t)m * DK + d] = (bf16)(s * qg[d]);
        Kb[(size_t)m * DK + d] = (bf16)(s * kg[d]);
      }
    }
  }
}

// ---------------------------------------------------------------------------
// Attention pass 1: per-row max and sum(exp) over all keys. 16 rows / wave.
// ---------------------------------------------------------------------------

__global__ __launch_bounds__(32) void gau_attn_stats(
    const bf16* __restrict__ Qb, const bf16* __restrict__ Kb,
    const int* __restrict__ mask, const float* __restrict__ scl,
    float* __restrict__ rowmax, float* __restrict__ rowsum) {
  const int qt = blockIdx.x;            // 512 query tiles of 16 rows
  const int b = qt >> 7;                // 128 tiles per batch
  const int m0 = qt * 16;               // global row index
  const int lane = threadIdx.x & 31, half = lane >> 4, lr = lane & 15;

  const bf16* Kbb = Kb + (size_t)b * SEQ * DK;
  const int* mkb = mask + b * SEQ;
  const float qs = scl[2 * b], neg = scl[2 * b + 1];

  v16bf qf[4];
#pragma unroll
  for (int kk = 0; kk < 4; ++kk) qf[kk] = frag_a(Qb, DK, m0, kk * 32);

  float vmax[8];
#pragma unroll
  for (int r = 0; r < 8; ++r) vmax[r] = -3.0e38f;

  for (int n0 = 0; n0 < SEQ; n0 += 16) {
    v8f c = {};
#pragma unroll
    for (int kk = 0; kk < 4; ++kk)
      c = wmma_bf16(qf[kk], frag_bt(Kbb, DK, n0, kk * 32), c);
    const int mk = mkb[n0 + lr];
#pragma unroll
    for (int r = 0; r < 8; ++r) {
      float s = mk ? c[r] * qs : neg;
      vmax[r] = fmaxf(vmax[r], s);
    }
  }
#pragma unroll
  for (int r = 0; r < 8; ++r)
    for (int o = 1; o < 16; o <<= 1)
      vmax[r] = fmaxf(vmax[r], __shfl_xor(vmax[r], o, 32));

  float vsum[8] = {};
  for (int n0 = 0; n0 < SEQ; n0 += 16) {
    v8f c = {};
#pragma unroll
    for (int kk = 0; kk < 4; ++kk)
      c = wmma_bf16(qf[kk], frag_bt(Kbb, DK, n0, kk * 32), c);
    const int mk = mkb[n0 + lr];
#pragma unroll
    for (int r = 0; r < 8; ++r) {
      float s = mk ? c[r] * qs : neg;
      vsum[r] += __expf(s - vmax[r]);
    }
  }
#pragma unroll
  for (int r = 0; r < 8; ++r)
    for (int o = 1; o < 16; o <<= 1) vsum[r] += __shfl_xor(vsum[r], o, 32);

  if (lr == 0) {
#pragma unroll
    for (int r = 0; r < 8; ++r) {
      rowmax[m0 + r + 8 * half] = vmax[r];
      rowsum[m0 + r + 8 * half] = vsum[r];
    }
  }
}

// ---------------------------------------------------------------------------
// Attention pass 2: G = u * (softmax(P) @ V) for a 16-row x 128-col tile.
// Scores recomputed, exp'd P staged through LDS (C-layout -> A-fragment).
// ---------------------------------------------------------------------------

__global__ __launch_bounds__(32) void gau_attn_pv(
    const bf16* __restrict__ Qb, const bf16* __restrict__ Kb,
    const bf16* __restrict__ Vt, const bf16* __restrict__ U,
    const int* __restrict__ mask, const float* __restrict__ scl,
    const float* __restrict__ rowmax, const float* __restrict__ rowsum,
    bf16* __restrict__ G) {
  __shared__ bf16 Plds[16 * 32];  // one 16x32 exp(score) tile, bf16

  const int qt = blockIdx.y;
  const int c0 = blockIdx.x * 128;
  const int b = qt >> 7, m0 = qt * 16;
  const int lane = threadIdx.x & 31, half = lane >> 4, lr = lane & 15;

  const bf16* Kbb = Kb + (size_t)b * SEQ * DK;
  const bf16* Vtb = Vt + (size_t)b * SEQ;  // column offset into key axis
  const int* mkb = mask + b * SEQ;
  const float qs = scl[2 * b], neg = scl[2 * b + 1];

  v16bf qf[4];
#pragma unroll
  for (int kk = 0; kk < 4; ++kk) qf[kk] = frag_a(Qb, DK, m0, kk * 32);

  float rm[8], rl[8];
#pragma unroll
  for (int r = 0; r < 8; ++r) {
    const int row = m0 + r + 8 * half;
    rm[r] = rowmax[row];
    rl[r] = 1.0f / fmaxf(rowsum[row], 1e-30f);
  }

  v8f acc[8] = {};

  for (int n0 = 0; n0 < SEQ; n0 += 32) {
    // scores for 32 keys -> exp -> LDS (two 16x16 tiles)
#pragma unroll
    for (int t = 0; t < 2; ++t) {
      const int nn = n0 + 16 * t;
      v8f c = {};
#pragma unroll
      for (int kk = 0; kk < 4; ++kk)
        c = wmma_bf16(qf[kk], frag_bt(Kbb, DK, nn, kk * 32), c);
      const int mk = mkb[nn + lr];
#pragma unroll
      for (int r = 0; r < 8; ++r) {
        float s = mk ? c[r] * qs : neg;
        Plds[(r + 8 * half) * 32 + 16 * t + lr] = (bf16)__expf(s - rm[r]);
      }
    }
    __syncthreads();
    v16bf pf = frag_a(Plds, 32, 0, 0);  // lane-transpose via LDS
    __syncthreads();

#pragma unroll
    for (int j = 0; j < 8; ++j) {
      __builtin_prefetch(Vtb + (size_t)(c0 + 16 * j + lr) * BS + n0 + 64, 0, 0);
      acc[j] = wmma_bf16(pf, frag_bt(Vtb, BS, c0 + 16 * j, n0), acc[j]);
    }
  }

#pragma unroll
  for (int j = 0; j < 8; ++j) {
    const int col = c0 + 16 * j + lr;
#pragma unroll
    for (int r = 0; r < 8; ++r) {
      const int row = m0 + r + 8 * half;
      float val = acc[j][r] * rl[r];
      val *= (float)U[(size_t)row * INTER + col];
      G[(size_t)row * INTER + col] = (bf16)val;
    }
  }
}

// ---------------------------------------------------------------------------
// GEMM2: O = G @ WoT^T, f32 output. 16x64 tile / wave.
// ---------------------------------------------------------------------------

__global__ __launch_bounds__(32) void gau_gemm2(
    const bf16* __restrict__ G, const bf16* __restrict__ WoT,
    float* __restrict__ O) {
  const int m0 = blockIdx.y * 16;
  const int n0 = blockIdx.x * 64;
  const int lane = threadIdx.x & 31, half = lane >> 4, lr = lane & 15;

  v8f acc[4] = {};
  for (int k0 = 0; k0 < INTER; k0 += 32) {
    v16bf a = frag_a(G, INTER, m0, k0);
#pragma unroll
    for (int j = 0; j < 4; ++j)
      acc[j] = wmma_bf16(a, frag_bt(WoT, INTER, n0 + 16 * j, k0), acc[j]);
  }
#pragma unroll
  for (int j = 0; j < 4; ++j) {
    const int col = n0 + 16 * j + lr;
#pragma unroll
    for (int r = 0; r < 8; ++r)
      O[(size_t)(m0 + r + 8 * half) * HID + col] = acc[j][r];
  }
}

// ---------------------------------------------------------------------------
// Host launcher
// ---------------------------------------------------------------------------

extern "C" void kernel_launch(void* const* d_in, const int* in_sizes, int n_in,
                              void* d_out, int out_size, void* d_ws, size_t ws_size,
                              hipStream_t stream) {
  const float* hs = (const float*)d_in[0];   // (4,2048,768)
  const float* Wi = (const float*)d_in[1];   // (768,3200)
  const float* Wo = (const float*)d_in[2];   // (1536,768)
  const float* qg = (const float*)d_in[3];   // (128,)
  const float* kg = (const float*)d_in[4];   // (128,)
  const int* msk  = (const int*)d_in[5];     // (4,2048)
  float* out = (float*)d_out;                // (4,2048,768)

  char* w = (char*)d_ws;
  auto carve = [&](size_t bytes) -> char* {
    char* p = w;
    w += (bytes + 255) & ~(size_t)255;
    return p;
  };
  bf16* Hb  = (bf16*)carve((size_t)BS * HID * 2);     // 12.6 MB
  bf16* WiT = (bf16*)carve((size_t)NI * HID * 2);     //  4.9 MB
  bf16* WoT = (bf16*)carve((size_t)HID * INTER * 2);  //  2.4 MB
  bf16* U   = (bf16*)carve((size_t)BS * INTER * 2);   // 25.2 MB
  bf16* Vt  = (bf16*)carve((size_t)INTER * BS * 2);   // 25.2 MB
  bf16* Qb  = (bf16*)carve((size_t)BS * DK * 2);      //  2.1 MB
  bf16* Kb  = (bf16*)carve((size_t)BS * DK * 2);      //  2.1 MB
  bf16* G   = (bf16*)carve((size_t)BS * INTER * 2);   // 25.2 MB
  float* rowmax = (float*)carve((size_t)BS * 4);
  float* rowsum = (float*)carve((size_t)BS * 4);
  float* scl    = (float*)carve(2 * NB * 4);

  // Prep: bf16 conversions / transposes / softmax scale
  gau_cvt_bf16<<<(BS * HID + 255) / 256, 256, 0, stream>>>(hs, Hb, BS * HID);
  gau_transpose_bf16<<<(HID * NI + 255) / 256, 256, 0, stream>>>(Wi, WiT, HID, NI);
  gau_transpose_bf16<<<(INTER * HID + 255) / 256, 256, 0, stream>>>(Wo, WoT, INTER, HID);
  gau_stats<<<NB, 256, 0, stream>>>(msk, scl);

  // GEMM1 + split/activation
  gau_gemm1<<<dim3(NI / 64, BS / 16), 32, 0, stream>>>(Hb, WiT, qg, kg, U, Vt, Qb, Kb);

  // Attention: stats then P@V (fused with u-gating)
  gau_attn_stats<<<BS / 16, 32, 0, stream>>>(Qb, Kb, msk, scl, rowmax, rowsum);
  gau_attn_pv<<<dim3(INTER / 128, BS / 16), 32, 0, stream>>>(Qb, Kb, Vt, U, msk, scl,
                                                             rowmax, rowsum, G);

  // Output projection
  gau_gemm2<<<dim3(HID / 64, BS / 16), 32, 0, stream>>>(G, WoT, out);
}
